// Crf_layer_4724464025891
// MI455X (gfx1250) — matrix operand order repaired and checked by reading, not compiled, not run on hardware
//
#include <hip/hip_runtime.h>
#include <hip/hip_bf16.h>
#include <stdint.h>

typedef __attribute__((ext_vector_type(16))) _Float16 v16h;
typedef __attribute__((ext_vector_type(8)))  float    v8f;

#define CRF_B 512
#define CRF_T 512
#define CRF_K 128
#define MT    16           // batch tile (WMMA M)
#define CHUNK 32           // backtrack chunk length
#define NCH   16           // ceil(511 / 32)
#define BPBYTES ((size_t)(CRF_T - 1) * CRF_B * CRF_K)   // u8 backpointers

// ---------------------------------------------------------------------------
// Sequence score: unary + binary along the target path, masked by lens.
// ---------------------------------------------------------------------------
__global__ void crf_seq_score(const float* __restrict__ inputs,
                              const int* __restrict__ targets,
                              const int* __restrict__ lens,
                              const float* __restrict__ trans,
                              float* __restrict__ score) {
    const int b = blockIdx.x;
    const int tid = threadIdx.x;               // 128 threads
    __shared__ float red[128];
    const int len = lens[b];
    float s = 0.0f;
    for (int p = tid; p < CRF_T; p += 128) {
        if (p < len) {
            const int tg = targets[b * CRF_T + p];
            s += inputs[((size_t)b * CRF_T + p) * CRF_K + tg];
            if (p >= 1) s += trans[targets[b * CRF_T + p - 1] * CRF_K + tg];
        }
    }
    red[tid] = s;
    __syncthreads();
    for (int off = 64; off > 0; off >>= 1) {
        if (tid < off) red[tid] += red[tid + off];
        __syncthreads();
    }
    if (tid == 0) score[b] = red[0];
}

// ---------------------------------------------------------------------------
// Forward pass: log-norm via WMMA (exp-domain matmul) + Viterbi (VALU max-plus)
// One block = 16 batches, 8 waves x 16-state tiles. T-1 sequential steps.
// ---------------------------------------------------------------------------
__global__ void __launch_bounds__(256) crf_forward(
    const float* __restrict__ inputs,
    const int* __restrict__ lens,
    const float* __restrict__ trans,
    float* __restrict__ lognorm,
    int* __restrict__ lastv,
    unsigned char* __restrict__ bp)
{
    __shared__ _Float16 trans_h[CRF_K * CRF_K];   // 32 KB  (viterbi reads)
    __shared__ float alpha[MT][CRF_K];            //  8 KB  (log-norm state)
    __shared__ float valpha[MT][CRF_K];           //  8 KB  (viterbi state)
    __shared__ _Float16 Af16[MT][CRF_K];          //  4 KB  (WMMA A panel)
    __shared__ float xbuf[MT][CRF_K];             //  8 KB  (async x[t] tile)
    __shared__ float maxpart[MT][16];             //  1 KB
    __shared__ float rowmax[MT];

    const int tid  = threadIdx.x;
    const int lane = tid & 31;
    const int w    = tid >> 5;                    // wave id 0..7 -> N tile
    const int batch0 = blockIdx.x * MT;

    // --- init: trans (f16) to LDS, alpha/valpha = x[0] ---
    for (int idx = tid; idx < CRF_K * CRF_K; idx += 256)
        trans_h[idx] = (_Float16)trans[idx];
    for (int idx = tid; idx < MT * CRF_K; idx += 256) {
        const int m = idx >> 7, j = idx & 127;
        const float v = inputs[((size_t)(batch0 + m) * CRF_T) * CRF_K + j];
        alpha[m][j] = v;
        valpha[m][j] = v;
    }

    // --- B fragments: exp(trans) in registers for entire T loop ---
    // ISA layout (16-bit B, K=32): N = lane&15, K = e + 8*((e>>3) + (lane>>4)).
    v16h bfrag[4];
#pragma unroll
    for (int c = 0; c < 4; ++c)
#pragma unroll
        for (int e = 0; e < 16; ++e)
            bfrag[c][e] = (_Float16)__expf(trans[(c * 32 + e + 8 * ((e >> 3) + (lane >> 4))) * CRF_K + w * 16 + (lane & 15)]);

    // C-matrix lane mapping: n = lane&15, m = r + 8*(lane>>4)
    const int n  = lane & 15;
    const int mh = (lane >> 4) * 8;
    const int j  = w * 16 + n;                    // my output state column
    int mylens[8];
#pragma unroll
    for (int r = 0; r < 8; ++r) mylens[r] = lens[batch0 + mh + r];

    __syncthreads();

    for (int t = 1; t < CRF_T; ++t) {
        // --- issue async copy of x[t] tile into LDS (consumed at end of step) ---
#pragma unroll
        for (int q = 0; q < 2; ++q) {
            const int c   = tid * 2 + q;          // 512 x 16B chunks
            const int m   = c >> 5;
            const int col = c & 31;
            const unsigned long long ga = (unsigned long long)(uintptr_t)
                (inputs + ((size_t)(batch0 + m) * CRF_T + t) * CRF_K + col * 4);
            const uint32_t ldsa = (uint32_t)(uintptr_t)(&xbuf[m][col * 4]);
            asm volatile("global_load_async_to_lds_b128 %0, %1, off"
                         :: "v"(ldsa), "v"(ga) : "memory");
        }

        // --- phase 1: per-batch-row max of alpha ---
        {
            const int m = tid >> 4, g = tid & 15;
            float mx = alpha[m][g * 8];
#pragma unroll
            for (int u = 1; u < 8; ++u) mx = fmaxf(mx, alpha[m][g * 8 + u]);
            maxpart[m][g] = mx;
        }
        __syncthreads();
        if (tid < MT) {
            float mx = maxpart[tid][0];
#pragma unroll
            for (int u = 1; u < 16; ++u) mx = fmaxf(mx, maxpart[tid][u]);
            rowmax[tid] = mx;
        }
        __syncthreads();

        // --- phase 2: A panel = exp(alpha - rowmax) as f16 ---
        for (int idx = tid; idx < MT * CRF_K; idx += 256) {
            const int m = idx >> 7, jj = idx & 127;
            Af16[m][jj] = (_Float16)__expf(alpha[m][jj] - rowmax[m]);
        }
        __syncthreads();

        // --- phase 3: 4x WMMA over K=128 ---
        v8f acc = {};
#pragma unroll
        for (int c = 0; c < 4; ++c) {
            v16h a;
#pragma unroll
            for (int e = 0; e < 16; ++e)
                a[e] = Af16[lane & 15][c * 32 + e + 8 * ((e >> 3) + (lane >> 4))];
            acc = __builtin_amdgcn_wmma_f32_16x16x32_f16(
                false, a, false, bfrag[c], (short)0, acc, false, false);
        }

        // --- phase 4: Viterbi max-plus + argmax (VALU) ---
        float best[8];
        int   bidx[8];
#pragma unroll
        for (int r = 0; r < 8; ++r) { best[r] = -3.4e38f; bidx[r] = 0; }
        for (int i = 0; i < CRF_K; ++i) {
            const float tj = (float)trans_h[i * CRF_K + j];
#pragma unroll
            for (int r = 0; r < 8; ++r) {
                const float v = valpha[mh + r][i] + tj;
                if (v > best[r]) { best[r] = v; bidx[r] = i; }
            }
        }

        // --- phase 5: wait for x[t], update states, store backpointers ---
        asm volatile("s_wait_asynccnt 0" ::: "memory");
        __syncthreads();
#pragma unroll
        for (int r = 0; r < 8; ++r) {
            const int  m    = mh + r;
            const bool live = (t < mylens[r]);
            const float xv  = xbuf[m][j];
            const float na  = live ? (__logf(acc[r]) + rowmax[m] + xv) : alpha[m][j];
            const float nv  = live ? (best[r] + xv) : valpha[m][j];
            const int   nb  = live ? bidx[r] : j;
            alpha[m][j]  = na;
            valpha[m][j] = nv;
            bp[((size_t)(t - 1) * CRF_B + (batch0 + m)) * CRF_K + j] = (unsigned char)nb;
        }
        __syncthreads();
    }

    // --- finalize: log-norm (logsumexp) and viterbi argmax per batch ---
    if (tid < MT) {
        const int m = tid;
        float mx = alpha[m][0];
        for (int jj = 1; jj < CRF_K; ++jj) mx = fmaxf(mx, alpha[m][jj]);
        float s = 0.0f;
        for (int jj = 0; jj < CRF_K; ++jj) s += __expf(alpha[m][jj] - mx);
        lognorm[batch0 + m] = __logf(s) + mx;

        float vmx = valpha[m][0];
        int am = 0;
        for (int jj = 1; jj < CRF_K; ++jj)
            if (valpha[m][jj] > vmx) { vmx = valpha[m][jj]; am = jj; }
        lastv[batch0 + m] = am;
    }
}

// ---------------------------------------------------------------------------
// Backtrack, parallelized by map composition over 32-step chunks.
// Phase 1: per (batch, chunk) compose the 32 backpointer maps in LDS.
// ---------------------------------------------------------------------------
__global__ void __launch_bounds__(128) crf_bt_chunkmap(
    const unsigned char* __restrict__ bp,
    unsigned char* __restrict__ chunkmap) {
    const int b = blockIdx.x, c = blockIdx.y, tid = threadIdx.x;
    __shared__ unsigned char tile[CHUNK][CRF_K];
    const int lo = c * CHUNK;
    const int nrows = (511 - lo < CHUNK) ? (511 - lo) : CHUNK;
    for (int r = tid >> 5; r < nrows; r += 4)
        ((uint32_t*)tile[r])[tid & 31] =
            ((const uint32_t*)(bp + ((size_t)(lo + r) * CRF_B + b) * CRF_K))[tid & 31];
    __syncthreads();
    int x = tid;                                   // one thread per start state
    for (int r = nrows - 1; r >= 0; --r) x = tile[r][x];
    chunkmap[((size_t)c * CRF_B + b) * CRF_K + tid] = (unsigned char)x;
}

// Phase 2: per batch, walk the 16 chunk maps to get each chunk's entry state.
__global__ void crf_bt_entries(const unsigned char* __restrict__ chunkmap,
                               const int* __restrict__ lastv,
                               unsigned char* __restrict__ entries,
                               float* __restrict__ pred) {
    const int b = blockIdx.x * blockDim.x + threadIdx.x;
    if (b >= CRF_B) return;
    int e = lastv[b];
    pred[(size_t)b * CRF_T + (CRF_T - 1)] = (float)e;   // tags[T-1] = argmax
    for (int c = NCH - 1; c >= 0; --c) {
        entries[(size_t)c * CRF_B + b] = (unsigned char)e;
        e = chunkmap[((size_t)c * CRF_B + b) * CRF_K + e];
    }
}

// Phase 3: per (batch, chunk) walk 32 steps in LDS from the entry state.
__global__ void __launch_bounds__(128) crf_bt_walk(
    const unsigned char* __restrict__ bp,
    const unsigned char* __restrict__ entries,
    float* __restrict__ pred) {
    const int b = blockIdx.x, c = blockIdx.y, tid = threadIdx.x;
    __shared__ unsigned char tile[CHUNK][CRF_K];
    const int lo = c * CHUNK;
    const int nrows = (511 - lo < CHUNK) ? (511 - lo) : CHUNK;
    for (int r = tid >> 5; r < nrows; r += 4)
        ((uint32_t*)tile[r])[tid & 31] =
            ((const uint32_t*)(bp + ((size_t)(lo + r) * CRF_B + b) * CRF_K))[tid & 31];
    __syncthreads();
    if (tid == 0) {
        int x = entries[(size_t)c * CRF_B + b];      // tags[hi+1]
        for (int r = nrows - 1; r >= 0; --r) {
            x = tile[r][x];                          // tags[lo+r]
            pred[(size_t)b * CRF_T + lo + r] = (float)x;
        }
    }
}

// ---------------------------------------------------------------------------
// Loss: sum_b -(score - lognorm)
// ---------------------------------------------------------------------------
__global__ void crf_loss(const float* __restrict__ score,
                         const float* __restrict__ lognorm,
                         float* __restrict__ out_loss) {
    __shared__ float red[256];
    const int tid = threadIdx.x;
    float s = 0.0f;
    for (int b = tid; b < CRF_B; b += 256) s += -(score[b] - lognorm[b]);
    red[tid] = s;
    __syncthreads();
    for (int off = 128; off > 0; off >>= 1) {
        if (tid < off) red[tid] += red[tid + off];
        __syncthreads();
    }
    if (tid == 0) out_loss[0] = red[0];
}

// ---------------------------------------------------------------------------
extern "C" void kernel_launch(void* const* d_in, const int* in_sizes, int n_in,
                              void* d_out, int out_size, void* d_ws, size_t ws_size,
                              hipStream_t stream) {
    const float* inputs  = (const float*)d_in[0];   // (B,T,K) f32
    const int*   targets = (const int*)d_in[1];     // (B,T) i32
    const int*   lens    = (const int*)d_in[2];     // (B,) i32
    const float* trans   = (const float*)d_in[3];   // (K,K) f32
    float* out = (float*)d_out;                     // predict (B*T) then loss (1)

    unsigned char* ws = (unsigned char*)d_ws;
    unsigned char* bp = ws;                         // 33.4 MB u8 backpointers
    float* lognorm = (float*)(ws + BPBYTES);
    float* score   = lognorm + CRF_B;
    int*   lastv   = (int*)(score + CRF_B);
    unsigned char* chunkmap = (unsigned char*)(lastv + CRF_B);      // 1 MB
    unsigned char* entries  = chunkmap + (size_t)NCH * CRF_B * CRF_K;

    crf_seq_score<<<CRF_B, 128, 0, stream>>>(inputs, targets, lens, trans, score);
    crf_forward<<<CRF_B / MT, 256, 0, stream>>>(inputs, lens, trans, lognorm, lastv, bp);
    crf_bt_chunkmap<<<dim3(CRF_B, NCH), 128, 0, stream>>>(bp, chunkmap);
    crf_bt_entries<<<(CRF_B + 255) / 256, 256, 0, stream>>>(chunkmap, lastv, entries, out);
    crf_bt_walk<<<dim3(CRF_B, NCH), 128, 0, stream>>>(bp, entries, out);
    crf_loss<<<1, 256, 0, stream>>>(score, lognorm, out + (size_t)CRF_B * CRF_T);
}
